// MultiScaleReadout_32401233281334
// MI455X (gfx1250) — matrix operand order repaired and checked by reading, not compiled, not run on hardware
//
#include <hip/hip_runtime.h>
#include <hip/hip_bf16.h>
#include <math.h>

// ---------------------------------------------------------------------------
// MultiScaleReadout for MI455X (gfx1250, wave32, WMMA)
//   mean/max/att pools: fp32 streaming (x read once, coalesced per-feature)
//   gate MLP + local MLP GEMMs: v_wmma_f32_16x16x32_bf16 (f32 accumulate)
//   B fragments staged in LDS; anti-LICM opaque offset keeps the per-use
//   ds_load_b128 in-loop instead of hoist-to-VGPR + scratch spill.
// ---------------------------------------------------------------------------

typedef __attribute__((ext_vector_type(16))) __bf16 v16bf;
typedef __attribute__((ext_vector_type(8)))  float  v8f;

#define NSEG 1024
#define DIM  256           // D
#define HID  128           // H (gate hidden)
#define LOC  128           // L (local pool dim)
#define OUTW (3*DIM + LOC) // 896

// workspace layout (bytes); total need ~2.28 MB
#define WS_OFF_STARTS 0
#define WS_OFF_W1P    16384
#define WS_OFF_LPP    (16384 + 131072)   // contiguous after w1p
#define WS_OFF_GATE   (16384 + 262144)

// per matrix: 64 frags * 32 lanes * 16 bf16 = 32768 bf16 = 64 KB; both = 128 KB
#define LDS_W_BYTES   131072

__device__ __forceinline__ float gelu_exact(float v) {
    return 0.5f * v * (1.0f + erff(v * 0.70710678118654752f));
}

// ---------------------------------------------------------------------------
// Kernel 0: segment start offsets via lower_bound over sorted batch[].
// ---------------------------------------------------------------------------
__global__ void seg_starts_kernel(const int* __restrict__ batch, int n,
                                  int* __restrict__ starts) {
    int g = blockIdx.x * blockDim.x + threadIdx.x;
    if (g > NSEG) return;
    if (g == NSEG) { starts[g] = n; return; }
    int lo = 0, hi = n;
    while (lo < hi) {
        int mid = (lo + hi) >> 1;
        if (batch[mid] < g) lo = mid + 1; else hi = mid;
    }
    starts[g] = lo;
}

// ---------------------------------------------------------------------------
// Kernel 1: convert w1 / lp_w (f32 [256][128] row-major) to bf16, pre-swizzled
// into dense-B WMMA fragment order (16x16x32 bf16):
//   frag(kt,nt): lane l (half h=l/16, col n=nt*16+l%16), element e in 0..15:
//     K = kt*32 + 16*h + e   -> contiguous 32B per lane.
// ---------------------------------------------------------------------------
__global__ void pack_weights_kernel(const float* __restrict__ w1,
                                    const float* __restrict__ lpw,
                                    __bf16* __restrict__ w1p,
                                    __bf16* __restrict__ lpp) {
    int p = blockIdx.x * blockDim.x + threadIdx.x;   // 0 .. 2*65536-1
    if (p >= 2 * 65536) return;
    int which = p >> 16;
    int q     = p & 65535;
    int frag  = q >> 9;          // kt*8 + nt
    int rem   = q & 511;
    int l     = rem >> 4;
    int e     = rem & 15;
    int kt = frag >> 3, nt = frag & 7;
    int h  = l >> 4;
    int K  = kt * 32 + 16 * h + e;
    int n  = nt * 16 + (l & 15);
    const float* src = which ? lpw : w1;
    __bf16*      dst = which ? lpp : w1p;
    dst[q] = (__bf16)src[K * HID + n];
}

// ---------------------------------------------------------------------------
// Kernel 2: one block (256 thr = 8 waves) per segment.
// Dynamic LDS: 131072 bytes holding both packed weight matrices
//   [0 .. 32767]  bf16  : w1 fragments (frag f, lane l, elem e)
//   [32768..65535] bf16 : lp_w fragments
// ---------------------------------------------------------------------------
__global__ __launch_bounds__(256)
void readout_kernel(const float* __restrict__ x,
                    const int*   __restrict__ starts,
                    const float* __restrict__ b1,
                    const float* __restrict__ w2,
                    const float* __restrict__ b2v,
                    const float* __restrict__ lp_b,
                    const __bf16* __restrict__ wpacked,   // w1p (lpp contiguous after)
                    float* __restrict__ gate_ws,
                    float* __restrict__ out) {
    extern __shared__ __bf16 s_wf[];          // 65536 bf16 = 128 KB
    __shared__ float s_sum_local[LOC];
    __shared__ float s_red[256];
    __shared__ float s_ebuf[1024];

    const int g    = blockIdx.x;
    const int tid  = threadIdx.x;
    const int wave = tid >> 5;
    const int lane = tid & 31;

    const int s   = starts[g];
    const int e   = starts[g + 1];
    const int cnt = e - s;

    // ---- stage both packed weight matrices into LDS (coalesced 16B copies)
    {
        const uint4* src = (const uint4*)wpacked;
        uint4*       dst = (uint4*)s_wf;
        for (int i = tid; i < LDS_W_BYTES / 16; i += 256)
            dst[i] = src[i];
    }
    if (tid < LOC) s_sum_local[tid] = 0.f;
    __syncthreads();

    // ---------------- Phase A: WMMA gate + local MLP, per-wave 16-node tiles
    const int nTiles = (cnt + 15) >> 4;
    const int m = lane & 15;       // A-matrix row within tile
    const int h = lane >> 4;       // lane half
    const int n = lane & 15;       // C/D column
    const float b2 = b2v[0];

    for (int t = wave; t < nTiles; t += 8) {
        // Opaque zero, re-materialized each iteration: defeats LICM on the
        // B-fragment LDS loads (otherwise the compiler hoists 512 VGPRs of
        // fragments out of the loop and spills them to scratch).
        int opq;
        asm volatile("s_mov_b32 %0, 0" : "=s"(opq));
        const __bf16* wf0 = s_wf + opq;            // w1 fragments
        const __bf16* wf1 = s_wf + 32768 + opq;    // lp_w fragments

        const int base  = s + t * 16;
        const int valid = min(16, cnt - t * 16);

        // Build A fragments (16x32 bf16 per k-tile). Layout: lane(h,m),
        // element e: K = kt*32 + 16h + e (matches packed B)
        v16bf a[8];
        if (m < valid) {
            const float* row = x + (size_t)(base + m) * DIM;
            #pragma unroll
            for (int kt = 0; kt < 8; ++kt) {
                const float* r0 = row + kt * 32 + 16 * h;
                const float* r1 = r0 + 8;
                #pragma unroll
                for (int i = 0; i < 8; ++i) a[kt][i]     = (__bf16)r0[i];
                #pragma unroll
                for (int i = 0; i < 8; ++i) a[kt][8 + i] = (__bf16)r1[i];
            }
        } else {
            #pragma unroll
            for (int kt = 0; kt < 8; ++kt)
                #pragma unroll
                for (int i = 0; i < 16; ++i) a[kt][i] = (__bf16)0.f;
        }

        // ---- gate MLP: gp[r] accumulates gelu(h)*w2 per (row r+8h, col n)
        float gp[8];
        #pragma unroll
        for (int r = 0; r < 8; ++r) gp[r] = 0.f;

        #pragma unroll
        for (int nt = 0; nt < 8; ++nt) {
            v8f c = {0.f, 0.f, 0.f, 0.f, 0.f, 0.f, 0.f, 0.f};
            #pragma unroll
            for (int kt = 0; kt < 8; ++kt) {
                v16bf b = *reinterpret_cast<const v16bf*>(
                    wf0 + (size_t)((kt * 8 + nt) * 32 + lane) * 16);
                c = __builtin_amdgcn_wmma_f32_16x16x32_bf16(
                        false, a[kt], false, b, (short)0, c, false, false);
            }
            const float bias = b1[nt * 16 + n];
            const float wv   = w2[nt * 16 + n];
            #pragma unroll
            for (int r = 0; r < 8; ++r)
                gp[r] += gelu_exact(c[r] + bias) * wv;
        }
        // reduce over the 16 lanes sharing each row group
        #pragma unroll
        for (int mask = 1; mask < 16; mask <<= 1)
            #pragma unroll
            for (int r = 0; r < 8; ++r)
                gp[r] += __shfl_xor(gp[r], mask, 16);
        if ((lane & 15) == 0) {
            #pragma unroll
            for (int r = 0; r < 8; ++r) {
                int rowi = r + 8 * h;
                if (rowi < valid) gate_ws[base + rowi] = gp[r] + b2;
            }
        }

        // ---- local MLP pool: sum over valid rows of gelu(x@lp_w + lp_b)
        #pragma unroll
        for (int nt = 0; nt < 8; ++nt) {
            v8f c = {0.f, 0.f, 0.f, 0.f, 0.f, 0.f, 0.f, 0.f};
            #pragma unroll
            for (int kt = 0; kt < 8; ++kt) {
                v16bf b = *reinterpret_cast<const v16bf*>(
                    wf1 + (size_t)((kt * 8 + nt) * 32 + lane) * 16);
                c = __builtin_amdgcn_wmma_f32_16x16x32_bf16(
                        false, a[kt], false, b, (short)0, c, false, false);
            }
            const float bias = lp_b[nt * 16 + n];
            float sl = 0.f;
            #pragma unroll
            for (int r = 0; r < 8; ++r) {
                int rowi = r + 8 * h;
                if (rowi < valid) sl += gelu_exact(c[r] + bias);
            }
            atomicAdd(&s_sum_local[nt * 16 + n], sl);   // ds_add_f32
        }
    }

    __threadfence();      // make gate_ws stores visible block-wide
    __syncthreads();

    // ---------------- Phase B: per-segment softmax stats over gates
    float lmax = -INFINITY;
    for (int i = tid; i < cnt; i += 256) lmax = fmaxf(lmax, gate_ws[s + i]);
    s_red[tid] = lmax;
    __syncthreads();
    for (int st = 128; st > 0; st >>= 1) {
        if (tid < st) s_red[tid] = fmaxf(s_red[tid], s_red[tid + st]);
        __syncthreads();
    }
    const float gmax = s_red[0];
    __syncthreads();

    float lsum = 0.f;
    for (int i = tid; i < cnt; i += 256) lsum += expf(gate_ws[s + i] - gmax);
    s_red[tid] = lsum;
    __syncthreads();
    for (int st = 128; st > 0; st >>= 1) {
        if (tid < st) s_red[tid] += s_red[tid + st];
        __syncthreads();
    }
    const float denom = s_red[0];
    __syncthreads();

    // ---------------- Phase C: fused streaming pass (mean + max + att)
    // thread = feature column -> each node row is one coalesced 1KB burst
    const int f = tid;
    float sumv = 0.f, maxv = -INFINITY, attv = 0.f;
    for (int c0 = 0; c0 < cnt; c0 += 1024) {
        const int nc = min(1024, cnt - c0);
        for (int i = tid; i < nc; i += 256)
            s_ebuf[i] = expf(gate_ws[s + c0 + i] - gmax);
        __syncthreads();
        const float* xp = x + (size_t)(s + c0) * DIM + f;
        for (int i = 0; i < nc; ++i) {
            float v = xp[(size_t)i * DIM];
            if (i + 8 < nc) __builtin_prefetch(xp + (size_t)(i + 8) * DIM, 0, 0);
            sumv += v;
            maxv  = fmaxf(maxv, v);
            attv  = fmaf(s_ebuf[i], v, attv);
        }
        __syncthreads();
    }

    const float cntf = (cnt > 0) ? (float)cnt : 1.f;
    float* og = out + (size_t)g * OUTW;
    og[f]           = sumv / cntf;
    og[DIM + f]     = (cnt > 0) ? maxv : -INFINITY;
    og[2 * DIM + f] = (denom > 0.f) ? (attv / denom) : 0.f;
    if (f < LOC)
        og[3 * DIM + f] = s_sum_local[f] / cntf;
}

// ---------------------------------------------------------------------------
extern "C" void kernel_launch(void* const* d_in, const int* in_sizes, int n_in,
                              void* d_out, int out_size, void* d_ws, size_t ws_size,
                              hipStream_t stream) {
    const float* x     = (const float*)d_in[0];
    const int*   batch = (const int*)  d_in[1];
    const float* w1    = (const float*)d_in[2];
    const float* b1    = (const float*)d_in[3];
    const float* w2    = (const float*)d_in[4];
    const float* b2    = (const float*)d_in[5];
    const float* lp_w  = (const float*)d_in[6];
    const float* lp_b  = (const float*)d_in[7];
    const int nNodes   = in_sizes[0] / DIM;

    char* ws = (char*)d_ws;
    int*    starts  = (int*)   (ws + WS_OFF_STARTS);
    __bf16* w1p     = (__bf16*)(ws + WS_OFF_W1P);
    __bf16* lpp     = (__bf16*)(ws + WS_OFF_LPP);
    float*  gate_ws = (float*) (ws + WS_OFF_GATE);

    seg_starts_kernel<<<(NSEG + 1 + 255) / 256, 256, 0, stream>>>(batch, nNodes, starts);
    pack_weights_kernel<<<(2 * 65536) / 256, 256, 0, stream>>>(w1, lp_w, w1p, lpp);
    readout_kernel<<<NSEG, 256, LDS_W_BYTES, stream>>>(x, starts, b1, w2, b2, lp_b,
                                                       w1p, gate_ws, (float*)d_out);
}
